// MultiHeadSelfAttention_6863357739067
// MI455X (gfx1250) — matrix operand order repaired and checked by reading, not compiled, not run on hardware
//
#include <hip/hip_runtime.h>

// ---------------- problem constants ----------------
#define SEQ    2048
#define NBATCH 4
#define NH     16
#define HD     64
#define DMODEL 1024

typedef __attribute__((ext_vector_type(16))) __bf16 v16bf;
typedef __attribute__((ext_vector_type(8)))  __bf16 v8bf;
typedef __attribute__((ext_vector_type(8)))  float  v8f;
typedef int v4i_gcc __attribute__((vector_size(16)));   // matches builtin prototype
typedef unsigned short u16;
typedef unsigned int   u32;

union BF8  { uint4 u; v8bf v; };
union FRAG { v16bf v; v8bf h[2]; };

#if __has_builtin(__builtin_amdgcn_global_load_async_to_lds_b128) && \
    __has_builtin(__builtin_amdgcn_s_wait_asynccnt)
#define HAVE_ASYNC_LDS 1
#else
#define HAVE_ASYNC_LDS 0
#endif

__device__ __forceinline__ u16 f2bf(float f) {
  union { float f; u32 u; } x; x.f = f;
  u32 r = x.u + 0x7FFFu + ((x.u >> 16) & 1u);   // round-to-nearest-even
  return (u16)(r >> 16);
}

__device__ __forceinline__ v16bf frag_from(const u16* p0, const u16* p1) {
  BF8 c0, c1;
  c0.u = *reinterpret_cast<const uint4*>(p0);
  c1.u = *reinterpret_cast<const uint4*>(p1);
  FRAG r; r.h[0] = c0.v; r.h[1] = c1.v;
  return r.v;
}

// A-fragment (16xK=32) of bf16 from row-major [row][k], ld elements.
// Lanes 0-15: row=rowbase+l, K={kb+0..7, kb+16..23}; lanes 16-31: K={kb+8..15, kb+24..31}.
__device__ __forceinline__ v16bf load_a(const u16* base, int rowbase, int ld, int kb,
                                        int lnib, int half) {
  const u16* p = base + (size_t)(rowbase + lnib) * ld + kb + half * 8;
  return frag_from(p, p + 16);
}

// B-fragment (K=32 x 16): column n = colbase+l taken from row-major source [col][k].
// Lanes 0-15: K={kb+0..7, kb+16..23}; lanes 16-31: K={kb+8..15, kb+24..31}.
__device__ __forceinline__ v16bf load_b(const u16* base, int colbase, int ld, int kb,
                                        int lnib, int half) {
  const u16* p = base + (size_t)(colbase + lnib) * ld + kb + half * 8;
  return frag_from(p, p + 16);
}

// ---- async global->LDS staging (ASYNCcnt path), with sync fallback ----
__device__ __forceinline__ void async_cp16(u16* lds, const u16* g) {
#if HAVE_ASYNC_LDS
  __builtin_amdgcn_global_load_async_to_lds_b128(
      (__attribute__((address_space(1))) v4i_gcc*)g,
      (__attribute__((address_space(3))) v4i_gcc*)lds, 0, 0);
#else
  *reinterpret_cast<uint4*>(lds) = *reinterpret_cast<const uint4*>(g);
#endif
}

template <int N>
__device__ __forceinline__ void wait_async() {
#if HAVE_ASYNC_LDS
  __builtin_amdgcn_s_wait_asynccnt(N);
#endif
}

// Stage a 64x64 bf16 tile: global row stride ldg -> LDS row stride 64.
// 512 chunks of 16B; 256 threads issue 2 each (2 async ops per thread).
__device__ __forceinline__ void stage_tile(u16* ldsbase, const u16* gbase, int ldg, int tid) {
#pragma unroll
  for (int i = 0; i < 2; ++i) {
    int idx = tid * 2 + i;                 // 0..511
    int row = idx >> 3, ch = (idx & 7) * 8;
    async_cp16(ldsbase + row * 64 + ch, gbase + (size_t)row * ldg + ch);
  }
}

// ---------------- fp32 -> bf16 pack ----------------
__global__ __launch_bounds__(256) void cvt_f32_bf16(const float* __restrict__ in,
                                                    u16* __restrict__ out, int n) {
  int i = blockIdx.x * blockDim.x + threadIdx.x;
  int stride = gridDim.x * blockDim.x;
  for (; i < n; i += stride) out[i] = f2bf(in[i]);
}

// ---------------- bf16 WMMA GEMM: C = A(MxK) * Bw^T, Bw is (Nc x K) row-major --------
// Wave tile 64x64 (4x4 WMMA accumulators), 2x4 waves -> WG tile 128x256.
// mode 0: bf16 row-major store
// mode 1: bf16 store scaled by 1/sqrt(HD) (Q projection)
// mode 2: bf16 store transposed per head -> Vt[(b*NH+h)*HD + dh][tok]
// mode 3: fp32 row-major store (output projection)
__global__ __launch_bounds__(256) void gemm_bf16_wmma(const u16* __restrict__ A,
                                                      const u16* __restrict__ Bw,
                                                      void* __restrict__ Cout,
                                                      int M, int Nc, int Kc, int mode) {
  const int tid  = threadIdx.x;
  const int lane = tid & 31, wid = tid >> 5;
  const int lnib = lane & 15, half = lane >> 4;
  const int wm = wid & 1, wn = wid >> 1;            // 2x4 wave grid
  const int mbase = blockIdx.x * 128 + wm * 64;     // 4 M-tiles per wave
  const int nbase = blockIdx.y * 256 + wn * 64;     // 4 N-tiles per wave
  (void)M;

  v8f acc[4][4];
#pragma unroll
  for (int i = 0; i < 4; ++i)
#pragma unroll
    for (int j = 0; j < 4; ++j) acc[i][j] = (v8f)(0.f);

  for (int k = 0; k < Kc; k += 32) {
    v16bf af[4], bfr[4];
#pragma unroll
    for (int i = 0; i < 4; ++i) af[i] = load_a(A, mbase + i * 16, Kc, k, lnib, half);
#pragma unroll
    for (int j = 0; j < 4; ++j) bfr[j] = load_b(Bw, nbase + j * 16, Kc, k, lnib, half);
#pragma unroll
    for (int i = 0; i < 4; ++i)
#pragma unroll
      for (int j = 0; j < 4; ++j)
        acc[i][j] = __builtin_amdgcn_wmma_f32_16x16x32_bf16(
            false, af[i], false, bfr[j], (short)0, acc[i][j], false, false);
  }

  // C/D layout: lane(0-15): n=lnib, m=r ; lane(16-31): n=lnib, m=r+8
#pragma unroll
  for (int i = 0; i < 4; ++i) {
#pragma unroll
    for (int j = 0; j < 4; ++j) {
#pragma unroll
      for (int r = 0; r < 8; ++r) {
        int m = mbase + i * 16 + half * 8 + r;
        int n = nbase + j * 16 + lnib;
        float v = acc[i][j][r];
        if (mode == 1) v *= 0.125f;                       // 1/sqrt(64)
        if (mode == 3) {
          ((float*)Cout)[(size_t)m * Nc + n] = v;
        } else if (mode == 2) {
          int b = m >> 11, tok = m & (SEQ - 1);
          int h = n >> 6,  dh  = n & (HD - 1);
          ((u16*)Cout)[((size_t)(b * NH + h) * HD + dh) * SEQ + tok] = f2bf(v);
        } else {
          ((u16*)Cout)[(size_t)m * Nc + n] = f2bf(v);
        }
      }
    }
  }
}

// ---------------- fused flash attention (per-head), bf16 WMMA, fp32 softmax ----------
// K/V 64-key blocks double-buffered in LDS via GLOBAL_LOAD_ASYNC_TO_LDS_B128.
__global__ __launch_bounds__(256) void attn_fmha(const u16* __restrict__ Qb,
                                                 const u16* __restrict__ Kb,
                                                 const u16* __restrict__ Vtb,
                                                 u16* __restrict__ AOb) {
  __shared__ __align__(16) u16 Kl[2][64 * 64];    // 2 x 8 KB : K block [key][d]
  __shared__ __align__(16) u16 Vl[2][64 * 64];    // 2 x 8 KB : V block [d][key]
  __shared__ __align__(16) u16 Pl[8 * 16 * 64];   // 16 KB    : per-wave P staging

  const int tid  = threadIdx.x;
  const int lane = tid & 31, wid = tid >> 5;
  const int lnib = lane & 15, half = lane >> 4;
  const int b = blockIdx.z, h = blockIdx.y;
  const int qbase = blockIdx.x * 128 + wid * 16;

  const u16* Qh  = Qb  + (size_t)b * SEQ * DMODEL + h * HD;
  const u16* Kh  = Kb  + (size_t)b * SEQ * DMODEL + h * HD;
  const u16* Vth = Vtb + (size_t)(b * NH + h) * HD * SEQ;
  u16*       AOh = AOb + (size_t)b * SEQ * DMODEL + h * HD;

  // Q fragments for this wave's 16 query rows (1/sqrt(Dh) already folded into Q)
  v16bf qf[2];
#pragma unroll
  for (int j = 0; j < 2; ++j) qf[j] = load_a(Qh, qbase, DMODEL, j * 32, lnib, half);

  v8f o[4];
#pragma unroll
  for (int d = 0; d < 4; ++d) o[d] = (v8f)(0.f);
  float m_run[8], l_run[8];
#pragma unroll
  for (int r = 0; r < 8; ++r) { m_run[r] = -1e30f; l_run[r] = 0.f; }

  u16* Pw = Pl + wid * (16 * 64);
  const int NBLK = SEQ / 64;

  // prologue: stage block 0 into buffer 0
  stage_tile(Kl[0], Kh + (size_t)0 * DMODEL, DMODEL, tid);
  stage_tile(Vl[0], Vth + 0, SEQ, tid);

  for (int kb = 0; kb < NBLK; ++kb) {
    const int key0 = kb * 64;
    const int cur = kb & 1, nxt = cur ^ 1;

    // overlap: issue next block's DMA, then wait for current block only
    if (kb + 1 < NBLK) {
      stage_tile(Kl[nxt], Kh + (size_t)(key0 + 64) * DMODEL, DMODEL, tid);
      stage_tile(Vl[nxt], Vth + (key0 + 64), SEQ, tid);
      wait_async<4>();
    } else {
      wait_async<0>();
    }
    __syncthreads();

    const u16* Kc = Kl[cur];
    const u16* Vc = Vl[cur];

    // ---- S = Q * K^T  (k-dim = Dh = 64 -> 2 steps; 4 key tiles) ----
    v8f s[4];
#pragma unroll
    for (int t = 0; t < 4; ++t) s[t] = (v8f)(0.f);
#pragma unroll
    for (int j = 0; j < 2; ++j) {
#pragma unroll
      for (int t = 0; t < 4; ++t) {
        v16bf kf = load_b(Kc, t * 16, 64, j * 32, lnib, half);
        s[t] = __builtin_amdgcn_wmma_f32_16x16x32_bf16(
            false, qf[j], false, kf, (short)0, s[t], false, false);
      }
    }

    // ---- online softmax (rows live in 16-lane halves; xor masks 1..8 stay inside) ----
    float mnew[8], alpha[8], rsum[8];
#pragma unroll
    for (int r = 0; r < 8; ++r) {
      float mb = fmaxf(fmaxf(s[0][r], s[1][r]), fmaxf(s[2][r], s[3][r]));
#pragma unroll
      for (int msk = 1; msk <= 8; msk <<= 1) mb = fmaxf(mb, __shfl_xor(mb, msk, 32));
      mnew[r]  = fmaxf(m_run[r], mb);
      alpha[r] = __expf(m_run[r] - mnew[r]);
      rsum[r]  = 0.f;
    }
#pragma unroll
    for (int t = 0; t < 4; ++t)
#pragma unroll
      for (int r = 0; r < 8; ++r) {
        float p = __expf(s[t][r] - mnew[r]);
        s[t][r] = p;
        rsum[r] += p;
      }
#pragma unroll
    for (int r = 0; r < 8; ++r) {
#pragma unroll
      for (int msk = 1; msk <= 8; msk <<= 1) rsum[r] += __shfl_xor(rsum[r], msk, 32);
      l_run[r] = l_run[r] * alpha[r] + rsum[r];
      m_run[r] = mnew[r];
    }
#pragma unroll
    for (int d = 0; d < 4; ++d)
#pragma unroll
      for (int r = 0; r < 8; ++r) o[d][r] *= alpha[r];

    // ---- P (C-layout) -> LDS row-major 16x64 bf16 ----
#pragma unroll
    for (int t = 0; t < 4; ++t)
#pragma unroll
      for (int r = 0; r < 8; ++r)
        Pw[(half * 8 + r) * 64 + t * 16 + lnib] = f2bf(s[t][r]);
    __syncthreads();

    // ---- O += P * V  (P A-frags + V B-frags from LDS) ----
#pragma unroll
    for (int j = 0; j < 2; ++j) {
      const u16* pp = Pw + lnib * 64 + j * 32 + half * 8;
      v16bf pf = frag_from(pp, pp + 16);
#pragma unroll
      for (int d = 0; d < 4; ++d) {
        v16bf vf = load_b(Vc, d * 16, 64, j * 32, lnib, half);
        o[d] = __builtin_amdgcn_wmma_f32_16x16x32_bf16(
            false, pf, false, vf, (short)0, o[d], false, false);
      }
    }
    // WAR guard: everyone must be done reading Kl/Vl[cur] and Pl before the
    // next iteration's async DMA / P writes land.
    __syncthreads();
  }

  // ---- normalize and store merged-head output (bf16) ----
#pragma unroll
  for (int d = 0; d < 4; ++d)
#pragma unroll
    for (int r = 0; r < 8; ++r) {
      int q = qbase + half * 8 + r;
      AOh[(size_t)q * DMODEL + d * 16 + lnib] = f2bf(o[d][r] / l_run[r]);
    }
}

// ---------------- host-side launch ----------------
extern "C" void kernel_launch(void* const* d_in, const int* in_sizes, int n_in,
                              void* d_out, int out_size, void* d_ws, size_t ws_size,
                              hipStream_t stream) {
  (void)in_sizes; (void)n_in; (void)out_size; (void)ws_size;
  const float* x  = (const float*)d_in[0];
  const float* Wq = (const float*)d_in[1];
  const float* Wk = (const float*)d_in[2];
  const float* Wv = (const float*)d_in[3];
  const float* Wo = (const float*)d_in[4];

  const size_t MT = (size_t)NBATCH * SEQ;           // 8192 token rows
  const size_t WE = (size_t)DMODEL * DMODEL;        // weight elems
  u16* xb  = (u16*)d_ws;                            // bf16 x           (16 MiB)
  u16* wqb = xb  + MT * DMODEL;                     // bf16 weights     (4x2 MiB)
  u16* wkb = wqb + WE;
  u16* wvb = wkb + WE;
  u16* wob = wvb + WE;
  u16* Qb  = wob + WE;                              // Q (scaled)       (16 MiB)
  u16* Kb  = Qb  + MT * DMODEL;                     // K                (16 MiB)
  u16* Vtb = Kb  + MT * DMODEL;                     // V transposed     (16 MiB)
  u16* AOb = Vtb + MT * DMODEL;                     // attn out         (16 MiB)

  const int nx = (int)(MT * DMODEL);
  const int nw = (int)WE;
  cvt_f32_bf16<<<512, 256, 0, stream>>>(x,  xb,  nx);
  cvt_f32_bf16<<<128, 256, 0, stream>>>(Wq, wqb, nw);
  cvt_f32_bf16<<<128, 256, 0, stream>>>(Wk, wkb, nw);
  cvt_f32_bf16<<<128, 256, 0, stream>>>(Wv, wvb, nw);
  cvt_f32_bf16<<<128, 256, 0, stream>>>(Wo, wob, nw);

  dim3 gg((unsigned)(MT / 128), DMODEL / 256);
  gemm_bf16_wmma<<<gg, 256, 0, stream>>>(xb, wqb, Qb,  (int)MT, DMODEL, DMODEL, 1);
  gemm_bf16_wmma<<<gg, 256, 0, stream>>>(xb, wkb, Kb,  (int)MT, DMODEL, DMODEL, 0);
  gemm_bf16_wmma<<<gg, 256, 0, stream>>>(xb, wvb, Vtb, (int)MT, DMODEL, DMODEL, 2);

  dim3 ga(SEQ / 128, NH, NBATCH);
  attn_fmha<<<ga, 256, 0, stream>>>(Qb, Kb, Vtb, AOb);

  gemm_bf16_wmma<<<gg, 256, 0, stream>>>(AOb, wob, d_out, (int)MT, DMODEL, DMODEL, 3);
}
